// Attention_87703232184528
// MI455X (gfx1250) — matrix operand order repaired and checked by reading, not compiled, not run on hardware
//
#include <hip/hip_runtime.h>
#include <stdint.h>

typedef __attribute__((ext_vector_type(16))) _Float16 v16h;
typedef __attribute__((ext_vector_type(8)))  _Float16 v8h;
typedef __attribute__((ext_vector_type(8)))  float    v8f;

#define BATCH 2
#define SEQ   2048
#define DM    1024
#define NH    16
#define HD    64

// ---------------------------------------------------------------------------
// CDNA5 async global->LDS copy (ASYNCcnt-tracked DMA), guarded so the code
// still compiles (register-staging fallback) on a toolchain without it.
// Builtin signature (from hipcc diagnostic): (v4i AS1*, v4i AS3*, imm, imm),
// where v4i = int __attribute__((vector_size(16))).
// ---------------------------------------------------------------------------
#if defined(__has_builtin)
#if __has_builtin(__builtin_amdgcn_global_load_async_to_lds_b128) && \
    __has_builtin(__builtin_amdgcn_s_wait_asynccnt)
#define ASYNC_COPY 1
#endif
#endif
#ifndef ASYNC_COPY
#define ASYNC_COPY 0
#endif

typedef int v4i __attribute__((vector_size(16)));
typedef __attribute__((address_space(1))) v4i gv4i;   // global
typedef __attribute__((address_space(3))) v4i lv4i;   // LDS

__device__ __forceinline__ void async_cp16(const _Float16* g, _Float16* l) {
#if ASYNC_COPY
  __builtin_amdgcn_global_load_async_to_lds_b128(
      (gv4i*)(uintptr_t)g,
      (lv4i*)(uint32_t)(uintptr_t)l,
      0, 0);
#else
  *(float4*)l = *(const float4*)g;   // sync fallback through VGPRs
#endif
}

__device__ __forceinline__ void async_wait_all() {
#if ASYNC_COPY
  __builtin_amdgcn_s_wait_asynccnt(0);
#endif
}

// ---------------------------------------------------------------------------
// Fragment loaders (wave32, v_wmma_f32_16x16x32_f16).
// A 16x32 (MxK): lane L holds M=L%16, K=(L/16)*8+{0..7} and +16+{0..7}.
// B 32x16 (KxN): lane L holds N=L%16, K=(L/16)*8+{0..7} and +16+{0..7}.
// Both are contiguous 16B chunks when tiles are row-major, k innermost.
// ---------------------------------------------------------------------------
__device__ __forceinline__ v16h ldsA16x32(const _Float16* base, int strideH) {
  const int lane = threadIdx.x & 31;
  const _Float16* p = base + (lane & 15) * strideH + ((lane >> 4) << 3);
  v8h lo = *(const v8h*)p;
  v8h hi = *(const v8h*)(p + 16);
  v16h r;
#pragma unroll
  for (int i = 0; i < 8; ++i) { r[i] = lo[i]; r[i + 8] = hi[i]; }
  return r;
}

__device__ __forceinline__ v16h ldsB32x16(const _Float16* base, int strideH) {
  const int lane = threadIdx.x & 31;
  const _Float16* p = base + (lane & 15) * strideH + ((lane >> 4) << 3);
  v8h lo = *(const v8h*)p;
  v8h hi = *(const v8h*)(p + 16);
  v16h r;
#pragma unroll
  for (int i = 0; i < 8; ++i) { r[i] = lo[i]; r[i + 8] = hi[i]; }
  return r;
}

// ---------------------------------------------------------------------------
// f32 -> f16 conversion (one-time)
// ---------------------------------------------------------------------------
__global__ void cvt_f32_f16(const float* __restrict__ in,
                            _Float16* __restrict__ out, int n) {
  int i = blockIdx.x * blockDim.x + threadIdx.x;
  int stride = gridDim.x * blockDim.x;
  for (; i < n; i += stride) out[i] = (_Float16)in[i];
}

// ---------------------------------------------------------------------------
// NT GEMM: C[M,N] = A[M,K] @ B[N,K]^T, f16 in, f32 accumulate.
// 128x128 block tile, 4 waves (2x2), 64x64 per wave => 16 WMMA : 16 ds_b128
// per k-step. Double-buffered LDS + async prefetch, ONE barrier per k-step.
// ---------------------------------------------------------------------------
#define GP 40  // padded LDS row stride in halfs (80B: 16B aligned, conflict-free)

__global__ __launch_bounds__(128) void gemm_nt_wmma(
    const _Float16* __restrict__ A, const _Float16* __restrict__ B,
    _Float16* __restrict__ outH, float* __restrict__ outF,
    const float* __restrict__ bias, float alpha, int M, int N, int K)
{
  __shared__ _Float16 sA[2][128 * GP];
  __shared__ _Float16 sB[2][128 * GP];

  const int tid   = threadIdx.x;   // 0..127
  const int lane  = tid & 31;
  const int wave  = tid >> 5;      // 0..3
  const int waveM = wave & 1;
  const int waveN = wave >> 1;
  const int mBase = blockIdx.x * 128;
  const int nBase = blockIdx.y * 128;

  v8f acc[4][4] = {};

  // each thread owns one 32-half row of each tile per buffer
  const _Float16* gA = A + (size_t)(mBase + tid) * K;
  const _Float16* gB = B + (size_t)(nBase + tid) * K;

  auto issue = [&](int k0, int buf) {
#pragma unroll
    for (int c = 0; c < 4; ++c) {
      async_cp16(gA + k0 + c * 8, &sA[buf][tid * GP + c * 8]);
      async_cp16(gB + k0 + c * 8, &sB[buf][tid * GP + c * 8]);
    }
  };

  issue(0, 0);
  const int nk = K >> 5;
  for (int i = 0; i < nk; ++i) {
    async_wait_all();
    __syncthreads();                       // publish buf[i&1]
    if (i + 1 < nk) issue((i + 1) << 5, (i + 1) & 1);  // overlap with compute

    const _Float16* cA = &sA[i & 1][0];
    const _Float16* cB = &sB[i & 1][0];
    v16h af[4], bf[4];
#pragma unroll
    for (int t = 0; t < 4; ++t) {
      af[t] = ldsA16x32(cA + (waveM * 64 + 16 * t) * GP, GP);
      bf[t] = ldsB32x16(cB + (waveN * 64 + 16 * t) * GP, GP);
    }
#pragma unroll
    for (int mi = 0; mi < 4; ++mi)
#pragma unroll
      for (int nj = 0; nj < 4; ++nj)
        acc[mi][nj] = __builtin_amdgcn_wmma_f32_16x16x32_f16(
            false, af[mi], false, bf[nj], (short)0, acc[mi][nj], false, false);
  }

  // C/D layout: lane L holds col n = L%16, rows m = 8*(L/16) + vgpr
  const int colOff = lane & 15;
  const int rowOff = (lane >> 4) << 3;
#pragma unroll
  for (int mi = 0; mi < 4; ++mi)
#pragma unroll
    for (int nj = 0; nj < 4; ++nj) {
      const int m0 = mBase + waveM * 64 + mi * 16 + rowOff;
      const int n0 = nBase + waveN * 64 + nj * 16 + colOff;
      if (outH) {
#pragma unroll
        for (int r = 0; r < 8; ++r)
          outH[(size_t)(m0 + r) * N + n0] = (_Float16)(acc[mi][nj][r] * alpha);
      } else {
        const float bv = bias ? bias[n0] : 0.f;
#pragma unroll
        for (int r = 0; r < 8; ++r)
          outF[(size_t)(m0 + r) * N + n0] = acc[mi][nj][r] * alpha + bv;
      }
    }
}

// ---------------------------------------------------------------------------
// Flash attention: grid (b*h, SEQ/128), 256 threads = 8 waves.
// Wave owns 16 query rows (Q fragments in registers); 32-key tiles,
// double-buffered K (async DMA) and V^T (register transpose), one barrier
// per tile. Q pre-scaled by hd^-0.5 at projection time.
// ---------------------------------------------------------------------------
#define KP 72  // K-tile LDS stride (144B: 16B aligned, conflict-free)
#define VP 40
#define PP 40

__global__ __launch_bounds__(256) void flash_attn_wmma(
    const _Float16* __restrict__ Q, const _Float16* __restrict__ Km,
    const _Float16* __restrict__ V, _Float16* __restrict__ O)
{
  __shared__ _Float16 sK[2][32 * KP];      // K tile, row-major [32 keys][64 hd]
  __shared__ _Float16 sVt[2][64 * VP];     // V tile transposed [64 hd][32 keys]
  __shared__ _Float16 sP[8 * 16 * PP];     // per-wave P tiles [16 rows][32 keys]

  const int tid  = threadIdx.x;
  const int lane = tid & 31;
  const int wave = tid >> 5;
  const int bh   = blockIdx.x;
  const int head = bh & 15;
  const int hOff = head * HD;
  const size_t rowBase = (size_t)(bh >> 4) * SEQ;
  const int qRow = blockIdx.y * 128 + wave * 16;

  // Q fragments: 16x64 = two 16x32 A fragments, pinned in registers
  v16h qf[2];
  {
    const _Float16* qp = Q + (rowBase + qRow + (lane & 15)) * DM + hOff
                           + ((lane >> 4) << 3);
#pragma unroll
    for (int s = 0; s < 2; ++s) {
      v8h lo = *(const v8h*)(qp + s * 32);
      v8h hi = *(const v8h*)(qp + s * 32 + 16);
#pragma unroll
      for (int i = 0; i < 8; ++i) { qf[s][i] = lo[i]; qf[s][i + 8] = hi[i]; }
    }
  }

  v8f oacc[4] = {};
  float mrow[8], lrow[8];
#pragma unroll
  for (int r = 0; r < 8; ++r) { mrow[r] = -3.0e38f; lrow[r] = 0.f; }

  const int ldJ = tid >> 3;        // 0..31 key row
  const int ldC = (tid & 7) << 3;  // 0..56 hd col
  const _Float16* gK = Km + (rowBase + ldJ) * DM + hOff + ldC;
  const _Float16* gV = V  + (rowBase + ldJ) * DM + hOff + ldC;

  v8h vreg;
  auto issueK  = [&](int jb, int buf) {
    async_cp16(gK + (size_t)jb * DM, &sK[buf][ldJ * KP + ldC]);
  };
  auto loadV   = [&](int jb) { vreg = *(const v8h*)(gV + (size_t)jb * DM); };
  auto commitV = [&](int buf) {
#pragma unroll
    for (int i2 = 0; i2 < 8; ++i2)
      sVt[buf][(ldC + i2) * VP + ldJ] = vreg[i2];
  };

  issueK(0, 0);
  loadV(0);
  commitV(0);

  const int nTiles = SEQ / 32;
  for (int i = 0; i < nTiles; ++i) {
    async_wait_all();
    __syncthreads();                        // publish buffers for tile i
    const int buf = i & 1;
    if (i + 1 < nTiles) { issueK((i + 1) * 32, buf ^ 1); loadV((i + 1) * 32); }

    // S(16x32) = Q(16x64) @ K^T(64x32): 2 col tiles x 2 k-steps
    v8f s0 = {}, s1 = {};
    {
      const _Float16* cK = &sK[buf][0];
      v16h b00 = ldsB32x16(cK + 0 * KP + 0,  KP);
      v16h b01 = ldsB32x16(cK + 0 * KP + 32, KP);
      v16h b10 = ldsB32x16(cK + 16 * KP + 0,  KP);
      v16h b11 = ldsB32x16(cK + 16 * KP + 32, KP);
      s0 = __builtin_amdgcn_wmma_f32_16x16x32_f16(false, qf[0], false, b00, (short)0, s0, false, false);
      s0 = __builtin_amdgcn_wmma_f32_16x16x32_f16(false, qf[1], false, b01, (short)0, s0, false, false);
      s1 = __builtin_amdgcn_wmma_f32_16x16x32_f16(false, qf[0], false, b10, (short)0, s1, false, false);
      s1 = __builtin_amdgcn_wmma_f32_16x16x32_f16(false, qf[1], false, b11, (short)0, s1, false, false);
    }

    // online softmax; xor masks 1..8 stay inside each 16-lane half (wave32)
    _Float16* myP = sP + wave * 16 * PP;
    const int rOff = (lane >> 4) << 3;
    const int cOff = lane & 15;
#pragma unroll
    for (int r = 0; r < 8; ++r) {
      float mx = fmaxf(s0[r], s1[r]);
#pragma unroll
      for (int off = 8; off >= 1; off >>= 1)
        mx = fmaxf(mx, __shfl_xor(mx, off, 32));
      const float mnew = fmaxf(mrow[r], mx);
      const float scl  = __expf(mrow[r] - mnew);
      const float p0   = __expf(s0[r] - mnew);
      const float p1   = __expf(s1[r] - mnew);
      float rs = p0 + p1;
#pragma unroll
      for (int off = 8; off >= 1; off >>= 1)
        rs += __shfl_xor(rs, off, 32);
      lrow[r] = lrow[r] * scl + rs;
      mrow[r] = mnew;
#pragma unroll
      for (int c = 0; c < 4; ++c) oacc[c][r] *= scl;
      myP[(rOff + r) * PP + cOff]      = (_Float16)p0;  // C-layout -> A-layout
      myP[(rOff + r) * PP + 16 + cOff] = (_Float16)p1;
    }

    // O(16x64) += P(16x32) @ V(32x64)
    v16h pf = ldsA16x32(myP, PP);
#pragma unroll
    for (int c = 0; c < 4; ++c) {
      v16h bv = ldsB32x16(&sVt[buf][(c * 16) * VP], VP);
      oacc[c] = __builtin_amdgcn_wmma_f32_16x16x32_f16(
          false, pf, false, bv, (short)0, oacc[c], false, false);
    }

    if (i + 1 < nTiles) commitV(buf ^ 1);   // after compute: overlapped V load
  }

  {  // epilogue: normalize and store f16
    const int rOff = (lane >> 4) << 3;
    const int cOff = lane & 15;
#pragma unroll
    for (int c = 0; c < 4; ++c)
#pragma unroll
      for (int r = 0; r < 8; ++r) {
        const float o = oacc[c][r] / lrow[r];
        O[(rowBase + qRow + rOff + r) * DM + hOff + c * 16 + cOff] = (_Float16)o;
      }
  }
}

// ---------------------------------------------------------------------------
// Host launch
// ---------------------------------------------------------------------------
extern "C" void kernel_launch(void* const* d_in, const int* in_sizes, int n_in,
                              void* d_out, int out_size, void* d_ws, size_t ws_size,
                              hipStream_t stream)
{
  const float* x  = (const float*)d_in[0];
  const float* wq = (const float*)d_in[1];
  const float* wk = (const float*)d_in[2];
  const float* wv = (const float*)d_in[3];
  const float* wo = (const float*)d_in[4];
  const float* bo = (const float*)d_in[5];

  const int rows = BATCH * SEQ;            // 4096
  const size_t xN = (size_t)rows * DM;     // 4Mi elems
  const size_t wN = (size_t)DM * DM;       // 1Mi elems

  char* p = (char*)d_ws;
  _Float16* xh  = (_Float16*)p; p += xN * 2;
  _Float16* wqh = (_Float16*)p; p += wN * 2;
  _Float16* wkh = (_Float16*)p; p += wN * 2;
  _Float16* wvh = (_Float16*)p; p += wN * 2;
  _Float16* woh = (_Float16*)p; p += wN * 2;
  _Float16* qh  = (_Float16*)p; p += xN * 2;
  _Float16* kh  = (_Float16*)p; p += xN * 2;
  _Float16* vh  = (_Float16*)p; p += xN * 2;
  _Float16* ah  = (_Float16*)p; p += xN * 2;   // total ~50.3 MB

  cvt_f32_f16<<<1024, 256, 0, stream>>>(x,  xh,  (int)xN);
  cvt_f32_f16<<<512,  256, 0, stream>>>(wq, wqh, (int)wN);
  cvt_f32_f16<<<512,  256, 0, stream>>>(wk, wkh, (int)wN);
  cvt_f32_f16<<<512,  256, 0, stream>>>(wv, wvh, (int)wN);
  cvt_f32_f16<<<512,  256, 0, stream>>>(wo, woh, (int)wN);

  const dim3 g(rows / 128, DM / 128);
  const float scale = 0.125f;  // HD^-0.5, folded into Q
  gemm_nt_wmma<<<g, 128, 0, stream>>>(xh, wqh, qh, nullptr, nullptr, scale, rows, DM, DM);
  gemm_nt_wmma<<<g, 128, 0, stream>>>(xh, wkh, kh, nullptr, nullptr, 1.0f, rows, DM, DM);
  gemm_nt_wmma<<<g, 128, 0, stream>>>(xh, wvh, vh, nullptr, nullptr, 1.0f, rows, DM, DM);

  flash_attn_wmma<<<dim3(BATCH * NH, SEQ / 128), 256, 0, stream>>>(qh, kh, vh, ah);

  gemm_nt_wmma<<<g, 128, 0, stream>>>(ah, woh, nullptr, (float*)d_out, bo, 1.0f, rows, DM, DM);
}
